// OrbitalMatrixMACE_11587821764683
// MI455X (gfx1250) — matrix operand order repaired and checked by reading, not compiled, not run on hardware
//
#include <hip/hip_runtime.h>
#include <hip/hip_bf16.h>

typedef __attribute__((ext_vector_type(16))) _Float16 v16h;
typedef __attribute__((ext_vector_type(8)))  _Float16 v8h;
typedef __attribute__((ext_vector_type(8)))  float    v8f;
typedef __attribute__((ext_vector_type(4)))  float    f4;
typedef int i4v __attribute__((vector_size(16)));

#define WMMA(a, b, c) __builtin_amdgcn_wmma_f32_16x16x32_f16(false, (a), false, (b), (short)0, (c), false, false)

#define NN 20000
#define EE 320000
#define CC 32

__device__ __constant__ int c_LOF[16] = {0,1,1,1,2,2,2,2,2,3,3,3,3,3,3,3};

// ---------------- optional async global->LDS staging (gfx1250) ----------------
#if defined(__has_builtin)
#if __has_builtin(__builtin_amdgcn_global_load_async_to_lds_b128)
#define HAVE_ASYNC_G2L 1
#endif
#endif
#ifndef HAVE_ASYNC_G2L
#define HAVE_ASYNC_G2L 0
#endif

#if HAVE_ASYNC_G2L
__device__ __forceinline__ void stage16(const float* g, float* l) {
  __builtin_amdgcn_global_load_async_to_lds_b128((i4v*)g, (i4v*)l, 0, 0);
}
__device__ __forceinline__ void stage_wait() {
#if __has_builtin(__builtin_amdgcn_s_wait_asynccnt)
  __builtin_amdgcn_s_wait_asynccnt(0);
#else
  asm volatile("s_wait_asynccnt 0x0" ::: "memory");
#endif
}
#else
__device__ __forceinline__ void stage16(const float* g, float* l) { *(f4*)l = *(const f4*)g; }
__device__ __forceinline__ void stage_wait() {}
#endif

// ---- WMMA fragment helpers (layouts per CDNA5 ISA, wave32) ----
// A (16xK=32 f16): lane row m = lane&15; halves 0..7 -> K=k0..k0+7,
// halves 8..15 -> K=k0+16..k0+23, k0 = (lane&16)>>1.

// A from f32, rows contiguous (32 floats), row stride rstride: 4x b128 loads
__device__ __forceinline__ v16h ldA_f32_row(const float* base, int rstride, int lane) {
  const float* row = base + (size_t)(lane & 15) * rstride;
  int k0 = (lane & 16) >> 1;               // 0 or 8
  const f4* p = (const f4*)row;
  f4 a0 = p[k0 >> 2], a1 = p[(k0 >> 2) + 1];
  f4 b0 = p[(k0 + 16) >> 2], b1 = p[((k0 + 16) >> 2) + 1];
  v16h a;
  #pragma unroll
  for (int i = 0; i < 4; ++i) {
    a[i]      = (_Float16)a0[i];
    a[4 + i]  = (_Float16)a1[i];
    a[8 + i]  = (_Float16)b0[i];
    a[12 + i] = (_Float16)b1[i];
  }
  return a;
}

// A from f16 rows (K=64 staging buffers): 2x 16B LDS loads
__device__ __forceinline__ v16h ldA_h_row(const _Float16* base, int rstride, int lane) {
  const _Float16* row = base + (size_t)(lane & 15) * rstride;
  int k0 = (lane & 16) >> 1;               // 0 or 8
  const v8h* p = (const v8h*)row;
  v8h lo = p[k0 >> 3];
  v8h hi = p[(k0 + 16) >> 3];
  v16h a;
  #pragma unroll
  for (int i = 0; i < 8; ++i) { a[i] = lo[i]; a[8 + i] = hi[i]; }
  return a;
}

// A from f32 16x8 tile zero-padded to K=32 (radial basis)
__device__ __forceinline__ v16h ldA_eb_row(const float* base, int lane) {
  v16h a = {};
  if ((lane & 16) == 0) {                  // k0==0: halves 0..7 are K=0..7, rest zero
    const f4* p = (const f4*)(base + (lane & 15) * 8);
    f4 a0 = p[0], a1 = p[1];
    #pragma unroll
    for (int i = 0; i < 4; ++i) { a[i] = (_Float16)a0[i]; a[4 + i] = (_Float16)a1[i]; }
  }
  return a;
}

// B (32x16 f16) from column-major LDS [ncols][K]: col j = lane&15,
// halves 0..7 -> K = k0..k0+7, halves 8..15 -> K = k0+16..k0+23, k0 = (lane&16)>>1  => 2x 16B LDS loads
__device__ __forceinline__ v16h ldB_col(const _Float16* colbase, int lane) {
  const v8h* p = (const v8h*)colbase;
  int kb = (lane & 16) >> 4;               // 0 or 1
  v8h lo = p[kb], hi = p[kb + 2];
  v16h b;
  #pragma unroll
  for (int i = 0; i < 8; ++i) { b[i] = lo[i]; b[8 + i] = hi[i]; }
  return b;
}

// D store: row = ((lane&16)>>1)+r, col = lane&15
__device__ __forceinline__ void stD(float* dst, int ld, int lane, v8f d) {
  int n = lane & 15; int rb = (lane & 16) >> 1;
  #pragma unroll
  for (int r = 0; r < 8; ++r) dst[(rb + r) * ld + n] = d[r];
}

__device__ __forceinline__ float silu_f(float x) { return x / (1.0f + __expf(-x)); }

// ---------------- Kernel A: geometry, spherical harmonics, bessel ----------------
__global__ void k_geom(const float* __restrict__ pos, const float* __restrict__ shifts,
                       const int* __restrict__ ei, float* __restrict__ Y,
                       float* __restrict__ eb) {
  int e = blockIdx.x * blockDim.x + threadIdx.x;
  if (e >= EE) return;
  int s = ei[e], d = ei[EE + e];
  float vx = pos[d*3+0] - pos[s*3+0] + shifts[e*3+0];
  float vy = pos[d*3+1] - pos[s*3+1] + shifts[e*3+1];
  float vz = pos[d*3+2] - pos[s*3+2] + shifts[e*3+2];
  float r = sqrtf(vx*vx + vy*vy + vz*vz) + 1e-9f;
  float x = vx / r, y = vy / r, z = vz / r;
  const float s3 = 1.73205080757f, s5 = 2.2360679775f, s15 = 3.87298334621f, s7 = 2.64575131106f;
  const float c358 = 2.09165006634f, c105 = 10.2469507660f, c218 = 1.62018517460f;
  float z2 = z*z, x2 = x*x, y2 = y*y;
  f4 sy0 = {1.0f, s3 * x, s3 * y, s3 * z};
  f4 sy1 = {s15 * x * y, s15 * y * z, 0.5f * s5 * (3.0f*z2 - 1.0f), s15 * x * z};
  f4 sy2 = {0.5f * s15 * (x2 - y2), c358 * y * (3.0f*x2 - y2), c105 * x * y * z,
            c218 * y * (5.0f*z2 - 1.0f)};
  f4 sy3 = {0.5f * s7 * z * (5.0f*z2 - 3.0f), c218 * x * (5.0f*z2 - 1.0f),
            0.5f * c105 * z * (x2 - y2), c358 * x * (x2 - 3.0f*y2)};
  f4* Ye = (f4*)(Y + (size_t)e * 16);
  Ye[0] = sy0; Ye[1] = sy1; Ye[2] = sy2; Ye[3] = sy3;
  // polynomial cutoff (P=5): f = 1 - 21 x^5 + 35 x^6 - 15 x^7
  float xc = r * 0.2f;
  float x5 = xc*xc*xc*xc*xc;
  float f = 1.0f - 21.0f*x5 + 35.0f*x5*xc - 15.0f*x5*xc*xc;
  f = (xc < 1.0f) ? f : 0.0f;
  const float cb = 0.632455532034f;   // sqrt(2/5)
  const float PI = 3.14159265358979f;
  f4 eb0, eb1;
  #pragma unroll
  for (int n = 1; n <= 8; ++n) {
    float v = cb * __sinf((float)n * PI * r * 0.2f) / r * f;
    if (n <= 4) eb0[n - 1] = v; else eb1[n - 5] = v;
  }
  f4* ep = (f4*)(eb + (size_t)e * 8);
  ep[0] = eb0; ep[1] = eb1;
}

// ---------------- Kernel B: node feature embedding (L=0 plane; rest pre-zeroed) ----------------
__global__ void k_embed(const float* __restrict__ attrs, const float* __restrict__ W_emb,
                        float* __restrict__ nf) {
  int idx = blockIdx.x * blockDim.x + threadIdx.x;
  if (idx >= NN * CC) return;
  int n = idx >> 5, c = idx & 31;
  float acc = 0.0f;
  #pragma unroll
  for (int k = 0; k < 10; ++k) acc += attrs[n*10 + k] * W_emb[k*32 + c];
  nf[(size_t)n * 512 + c] = acc;   // layout (N, L, C); L=0 plane
}

// ---------------- Kernel C: h0 = nf[:,:,L=0] @ W_up  (WMMA) ----------------
__global__ void k_h0(const float* __restrict__ nf, const float* __restrict__ W_up_t,
                     float* __restrict__ h0, int numTiles) {
  __shared__ __align__(16) _Float16 sW[32*32];        // transposed [col][k]
  for (int i = threadIdx.x; i < 1024; i += blockDim.x) {
    int j = i >> 5, k = i & 31;
    sW[j*32 + k] = (_Float16)W_up_t[k*32 + j];
  }
  __syncthreads();
  int lane = threadIdx.x & 31;
  int tile = blockIdx.x * (blockDim.x >> 5) + (threadIdx.x >> 5);
  if (tile >= numTiles) return;
  int n0 = tile * 16;
  v16h a = ldA_f32_row(nf + (size_t)n0 * 512, 512, lane);
  #pragma unroll
  for (int nt = 0; nt < 2; ++nt) {
    v8f z = {};
    v8f d = WMMA(a, ldB_col(sW + (nt*16 + (lane & 15))*32, lane), z);
    stD(h0 + (size_t)n0*32 + nt*16, 32, lane, d);
  }
}

// ---------------- Kernel D: radial MLP (WMMA) + message scatter ----------------
__global__ void k_msg(const float* __restrict__ eb, const float* __restrict__ Y,
                      const int* __restrict__ ei, const float* __restrict__ h0,
                      const float* __restrict__ rW1, const float* __restrict__ rb1,
                      const float* __restrict__ rW2, const float* __restrict__ rb2,
                      const float* __restrict__ rW3, float* __restrict__ agg,
                      int numTiles) {
  __shared__ __align__(16) _Float16 sW1[64*32];    // [j][k], k<8 valid
  __shared__ __align__(16) _Float16 sW2[64*64];    // [j][k]
  __shared__ __align__(16) _Float16 sW3[128*64];   // [j][k]
  __shared__ float sb1[64], sb2[64];
  __shared__ __align__(16) _Float16 xT[2][16*64];
  __shared__ __align__(16) float    rT[2][16*128];
  __shared__ __align__(16) float    ebT[2][16*8];
  __shared__ __align__(16) float    yT[2][16*16];
  for (int i = threadIdx.x; i < 64*32; i += blockDim.x) {
    int j = i >> 5, k = i & 31;
    sW1[i] = (k < 8) ? (_Float16)rW1[k*64 + j] : (_Float16)0.0f;
  }
  for (int i = threadIdx.x; i < 64*64; i += blockDim.x) {
    int j = i >> 6, k = i & 63;
    sW2[i] = (_Float16)rW2[k*64 + j];
  }
  for (int i = threadIdx.x; i < 128*64; i += blockDim.x) {
    int j = i >> 6, k = i & 63;
    sW3[i] = (_Float16)rW3[k*128 + j];
  }
  for (int i = threadIdx.x; i < 64; i += blockDim.x) { sb1[i] = rb1[i]; sb2[i] = rb2[i]; }
  __syncthreads();
  int w = threadIdx.x >> 5, lane = threadIdx.x & 31;
  int tile = blockIdx.x * 2 + w;
  if (tile >= numTiles) return;
  int e0 = tile * 16;
  _Float16* xw = xT[w];
  float*    rw = rT[w];
  float*    ebw = ebT[w];
  float*    yw  = yT[w];
  int col = lane & 15, rb = (lane & 16) >> 1;

  // async-stage eb (512B) and Y (1KB) tiles into LDS
  stage16(eb + (size_t)e0*8 + lane*4, ebw + lane*4);
  stage16(Y + (size_t)e0*16 + lane*4, yw + lane*4);
  stage16(Y + (size_t)e0*16 + 128 + lane*4, yw + 128 + lane*4);

  // edge endpoints + prefetch gather targets while MLP runs
  int e  = e0 + (lane & 15);
  int s  = ei[e], dd = ei[EE + e];
  __builtin_prefetch(h0 + (size_t)s*32, 0, 3);
  __builtin_prefetch(agg + (size_t)dd*512, 1, 3);

  stage_wait();

  // layer 1: (16x8 pad 32) @ (32x64) + bias, silu
  v16h aeb = ldA_eb_row(ebw, lane);
  #pragma unroll
  for (int nt = 0; nt < 4; ++nt) {
    v8f z = {};
    v8f d = WMMA(aeb, ldB_col(sW1 + (nt*16 + col)*32, lane), z);
    int j = nt*16 + col;
    float b = sb1[j];
    #pragma unroll
    for (int r = 0; r < 8; ++r) xw[(rb + r)*64 + j] = (_Float16)silu_f(d[r] + b);
  }
  // layer 2: (16x64) @ (64x64), K split in 2
  v16h a0 = ldA_h_row(xw, 64, lane);
  v16h a1 = ldA_h_row(xw + 32, 64, lane);
  v8f d2[4];
  #pragma unroll
  for (int nt = 0; nt < 4; ++nt) {
    v8f z = {};
    v8f d = WMMA(a0, ldB_col(sW2 + (nt*16 + col)*64, lane), z);
    d = WMMA(a1, ldB_col(sW2 + (nt*16 + col)*64 + 32, lane), d);
    d2[nt] = d;
  }
  #pragma unroll
  for (int nt = 0; nt < 4; ++nt) {
    int j = nt*16 + col;
    float b = sb2[j];
    #pragma unroll
    for (int r = 0; r < 8; ++r) xw[(rb + r)*64 + j] = (_Float16)silu_f(d2[nt][r] + b);
  }
  // layer 3: R = (16x64) @ (64x128)
  v16h b0 = ldA_h_row(xw, 64, lane);
  v16h b1 = ldA_h_row(xw + 32, 64, lane);
  #pragma unroll
  for (int nt = 0; nt < 8; ++nt) {
    v8f z = {};
    v8f d = WMMA(b0, ldB_col(sW3 + (nt*16 + col)*64, lane), z);
    d = WMMA(b1, ldB_col(sW3 + (nt*16 + col)*64 + 32, lane), d);
    stD(rw + nt*16, 128, lane, d);
  }
  // scatter: agg[dst, L, c] += R[e, L_OF[L]*32+c] * h0[src,c] * Y[e,L] / 16
  int ch = (lane >> 4) * 16;
  float h0v[16];
  const f4* hp = (const f4*)(h0 + (size_t)s*32 + ch);
  #pragma unroll
  for (int q = 0; q < 4; ++q) {
    f4 v = hp[q];
    #pragma unroll
    for (int j = 0; j < 4; ++j) h0v[q*4 + j] = v[j];
  }
  const float* rrow = rw + (lane & 15) * 128;
  float* aggRow = agg + (size_t)dd * 512;
  for (int L = 0; L < 16; ++L) {
    float yv = yw[(lane & 15)*16 + L] * 0.0625f;
    const float* rr = rrow + c_LOF[L]*32 + ch;
    float* ap = aggRow + L*32 + ch;
    #pragma unroll
    for (int ci = 0; ci < 16; ++ci) atomicAdd(ap + ci, rr[ci] * h0v[ci] * yv);
  }
}

// ---------------- Kernel E: node update (per-L WMMA mixes + gates) + node head ----------------
__global__ void k_node(const float* __restrict__ nf_in, const float* __restrict__ agg,
                       const float* __restrict__ W_lmix, const float* __restrict__ W_scm,
                       const float* __restrict__ W_hid, const float* __restrict__ g_sc,
                       const float* __restrict__ W_prod, const float* __restrict__ W_node,
                       const int* __restrict__ types, float* __restrict__ nf_out,
                       float* __restrict__ node_contrib, int t, int numTiles) {
  __shared__ __align__(16) _Float16 sLmix[4*32*32];  // [lof][j][k] transposed
  __shared__ __align__(16) _Float16 sScm[4*32*32];
  __shared__ __align__(16) _Float16 sHid[4*32*32];
  __shared__ __align__(16) _Float16 sWn[112*32];     // [j][k]
  __shared__ __align__(16) float    pfT[8][16*32];
  for (int i = threadIdx.x; i < 4096; i += blockDim.x) {
    int lof = i >> 10, j = (i >> 5) & 31, k = i & 31;
    int src = lof*1024 + k*32 + j;
    sLmix[i] = (_Float16)W_lmix[src];
    sScm[i]  = (_Float16)W_scm[src];
    sHid[i]  = (_Float16)W_hid[src];
  }
  for (int i = threadIdx.x; i < 112*32; i += blockDim.x) {
    int j = i >> 5, k = i & 31;
    sWn[i] = (j < 100) ? (_Float16)W_node[k*100 + j] : (_Float16)0.0f;
  }
  __syncthreads();
  int w = threadIdx.x >> 5, lane = threadIdx.x & 31;
  int tile = blockIdx.x * 8 + w;
  if (tile >= numTiles) return;
  int n0 = tile * 16;
  int col = lane & 15, rb = (lane & 16) >> 1;
  float* pw = pfT[w];

  int ty[8];
  #pragma unroll
  for (int r = 0; r < 8; ++r) ty[r] = types[n0 + rb + r];
  v8f gscf[2], w0f[2], w1f[2], w2f[2];
  #pragma unroll
  for (int nt = 0; nt < 2; ++nt) {
    int k = nt*16 + col;
    #pragma unroll
    for (int r = 0; r < 8; ++r) {
      gscf[nt][r] = g_sc[ty[r]*32 + k];
      const float* wp = W_prod + (size_t)ty[r] * 96;   // (3, C)
      w0f[nt][r] = wp[k]; w1f[nt][r] = wp[32 + k]; w2f[nt][r] = wp[64 + k];
    }
  }

  v8f gatef[2] = {}, scf[2] = {};
  for (int L = 0; L < 16; ++L) {
    int lof = c_LOF[L];
    v16h aggA = ldA_f32_row(agg  + (size_t)n0*512 + L*32, 512, lane);
    v16h nfA  = ldA_f32_row(nf_in + (size_t)n0*512 + L*32, 512, lane);
    #pragma unroll
    for (int nt = 0; nt < 2; ++nt) {
      v8f z = {};
      v8f mfr = WMMA(aggA, ldB_col(sLmix + lof*1024 + (nt*16 + col)*32, lane), z);
      v8f sc  = WMMA(nfA,  ldB_col(sScm  + lof*1024 + (nt*16 + col)*32, lane), z);
      #pragma unroll
      for (int r = 0; r < 8; ++r) sc[r] *= gscf[nt][r];
      scf[nt] = sc;
      if (L == 0) {
        #pragma unroll
        for (int r = 0; r < 8; ++r) {
          float s = mfr[r];
          gatef[nt][r] = w0f[nt][r] + s * w1f[nt][r] + s * s * w2f[nt][r];
        }
      }
      #pragma unroll
      for (int r = 0; r < 8; ++r) pw[(rb + r)*32 + nt*16 + col] = mfr[r] * gatef[nt][r];
    }
    v16h pfA = ldA_f32_row(pw, 32, lane);
    #pragma unroll
    for (int nt = 0; nt < 2; ++nt) {
      v8f d = WMMA(pfA, ldB_col(sHid + lof*1024 + (nt*16 + col)*32, lane), scf[nt]);
      #pragma unroll
      for (int r = 0; r < 8; ++r)
        nf_out[(size_t)(n0 + rb + r)*512 + L*32 + nt*16 + col] = d[r];
    }
  }
  // node head: nf_out[:, L=0, :] @ W_node (100 cols padded to 112)
  v16h ha = ldA_f32_row(nf_out + (size_t)n0*512, 512, lane);
  #pragma unroll
  for (int jt = 0; jt < 7; ++jt) {
    v8f z = {};
    v8f d = WMMA(ha, ldB_col(sWn + (jt*16 + col)*32, lane), z);
    int j = jt*16 + col;
    if (j < 100) {
      #pragma unroll
      for (int r = 0; r < 8; ++r)
        node_contrib[(size_t)(n0 + rb + r)*200 + j*2 + t] = d[r];
    }
  }
}

// ---------------- Kernel F: edge readout ----------------
__global__ void k_edge(const float* __restrict__ eb, const float* __restrict__ Y,
                       const int* __restrict__ ei, const float* __restrict__ nf,
                       const float* __restrict__ W_er, const float* __restrict__ W_edge,
                       float* __restrict__ edge_contrib, int t, int numTiles) {
  __shared__ __align__(16) _Float16 sWer[32*32];   // [j][k], k<8 valid
  __shared__ __align__(16) _Float16 sWe[112*32];   // [j][k]
  __shared__ __align__(16) float ReT[4][16*32];
  __shared__ __align__(16) float ivT[4][16*32];
  __shared__ __align__(16) float ebT[4][16*8];
  __shared__ __align__(16) float yT[4][16*16];
  for (int i = threadIdx.x; i < 1024; i += blockDim.x) {
    int j = i >> 5, k = i & 31;
    sWer[i] = (k < 8) ? (_Float16)W_er[k*32 + j] : (_Float16)0.0f;
  }
  for (int i = threadIdx.x; i < 112*32; i += blockDim.x) {
    int j = i >> 5, k = i & 31;
    sWe[i] = (j < 100) ? (_Float16)W_edge[k*100 + j] : (_Float16)0.0f;
  }
  __syncthreads();
  int w = threadIdx.x >> 5, lane = threadIdx.x & 31;
  int tile = blockIdx.x * 4 + w;
  if (tile >= numTiles) return;
  int e0 = tile * 16;
  float* Re = ReT[w];
  float* iv = ivT[w];
  float* ebw = ebT[w];
  float* yw = yT[w];
  int col = lane & 15, rb = (lane & 16) >> 1;

  stage16(eb + (size_t)e0*8 + lane*4, ebw + lane*4);
  stage16(Y + (size_t)e0*16 + lane*4, yw + lane*4);
  stage16(Y + (size_t)e0*16 + 128 + lane*4, yw + 128 + lane*4);

  int e = e0 + (lane & 15);
  int s = ei[e], dd = ei[EE + e];
  __builtin_prefetch(nf + (size_t)s*512, 0, 3);
  __builtin_prefetch(nf + (size_t)dd*512, 0, 3);

  stage_wait();

  // Re = silu(eb @ W_er)
  v16h aeb = ldA_eb_row(ebw, lane);
  #pragma unroll
  for (int nt = 0; nt < 2; ++nt) {
    v8f z = {};
    v8f d = WMMA(aeb, ldB_col(sWer + (nt*16 + col)*32, lane), z);
    #pragma unroll
    for (int r = 0; r < 8; ++r) Re[(rb + r)*32 + nt*16 + col] = silu_f(d[r]);
  }
  // inv[e,c] = sum_L (nf[src]+nf[dst])[L,c] * Y[e,L]; times Re
  int ch = (lane >> 4) * 16;
  const float* ps = nf + (size_t)s  * 512;
  const float* pd = nf + (size_t)dd * 512;
  float acc[16];
  #pragma unroll
  for (int i = 0; i < 16; ++i) acc[i] = 0.0f;
  for (int L = 0; L < 16; ++L) {
    float yv = yw[(lane & 15)*16 + L];
    const f4* pls = (const f4*)(ps + L*32 + ch);
    const f4* pld = (const f4*)(pd + L*32 + ch);
    #pragma unroll
    for (int q = 0; q < 4; ++q) {
      f4 a = pls[q], b = pld[q];
      #pragma unroll
      for (int j = 0; j < 4; ++j) acc[q*4 + j] += (a[j] + b[j]) * yv;
    }
  }
  #pragma unroll
  for (int ci = 0; ci < 16; ++ci) {
    int c = ch + ci;
    iv[(lane & 15)*32 + c] = acc[ci] * Re[(lane & 15)*32 + c];
  }
  // edge head: (16x32) @ (32x112)
  v16h ia = ldA_f32_row(iv, 32, lane);
  #pragma unroll
  for (int jt = 0; jt < 7; ++jt) {
    v8f z = {};
    v8f d = WMMA(ia, ldB_col(sWe + (jt*16 + col)*32, lane), z);
    int j = jt*16 + col;
    if (j < 100) {
      #pragma unroll
      for (int r = 0; r < 8; ++r)
        edge_contrib[(size_t)(e0 + rb + r)*200 + j*2 + t] = d[r];
    }
  }
}

// ---------------- Kernel G: sum over T ----------------
__global__ void k_sum(const float* __restrict__ contrib, float* __restrict__ sum, size_t n) {
  size_t i = (size_t)blockIdx.x * blockDim.x + threadIdx.x;
  if (i < n) sum[i] = contrib[i*2] + contrib[i*2 + 1];
}

extern "C" void kernel_launch(void* const* d_in, const int* in_sizes, int n_in,
                              void* d_out, int out_size, void* d_ws, size_t ws_size,
                              hipStream_t stream) {
  (void)in_sizes; (void)n_in; (void)out_size; (void)ws_size;
  const float* positions  = (const float*)d_in[0];
  const float* node_attrs = (const float*)d_in[1];
  const float* shifts     = (const float*)d_in[2];
  const float* W_emb      = (const float*)d_in[3];
  const float* rW1        = (const float*)d_in[4];
  const float* rb1        = (const float*)d_in[5];
  const float* rW2        = (const float*)d_in[6];
  const float* rb2        = (const float*)d_in[7];
  const float* rW3        = (const float*)d_in[8];
  const float* W_up       = (const float*)d_in[9];
  const float* W_lmix     = (const float*)d_in[10];
  const float* W_scm      = (const float*)d_in[11];
  const float* g_sc       = (const float*)d_in[12];
  const float* W_prod     = (const float*)d_in[13];
  const float* W_hid      = (const float*)d_in[14];
  const float* W_node     = (const float*)d_in[15];
  const float* W_edge     = (const float*)d_in[16];
  const float* W_er       = (const float*)d_in[17];
  const int*   edge_index = (const int*)d_in[18];
  const int*   node_types = (const int*)d_in[19];

  float* out = (float*)d_out;
  float* node_sum     = out;
  float* node_contrib = out + (size_t)NN * 100;
  float* edge_sum     = out + (size_t)NN * 300;
  float* edge_contrib = out + (size_t)NN * 300 + (size_t)EE * 100;

  float* ws  = (float*)d_ws;
  float* Y   = ws;                            // E*16
  float* eb  = Y   + (size_t)EE * 16;         // E*8
  float* nfA = eb  + (size_t)EE * 8;          // N*512 (layout N, L, C)
  float* nfB = nfA + (size_t)NN * 512;        // N*512
  float* h0  = nfB + (size_t)NN * 512;        // N*32
  float* agg = h0  + (size_t)NN * 32;         // N*512 (layout N, L, C)

  const int nodeTiles = NN / 16;   // 1250
  const int edgeTiles = EE / 16;   // 20000

  k_geom<<<(EE + 255) / 256, 256, 0, stream>>>(positions, shifts, edge_index, Y, eb);
  (void)hipMemsetAsync(nfA, 0, (size_t)NN * 512 * sizeof(float), stream);
  k_embed<<<(NN * CC + 255) / 256, 256, 0, stream>>>(node_attrs, W_emb, nfA);

  float* nf_in = nfA;
  float* nf_out = nfB;
  for (int t = 0; t < 2; ++t) {
    k_h0<<<(nodeTiles + 7) / 8, 256, 0, stream>>>(nf_in, W_up + t * 1024, h0, nodeTiles);
    (void)hipMemsetAsync(agg, 0, (size_t)NN * 512 * sizeof(float), stream);
    k_msg<<<(edgeTiles + 1) / 2, 64, 0, stream>>>(eb, Y, edge_index, h0,
        rW1 + t * 512, rb1 + t * 64, rW2 + t * 4096, rb2 + t * 64, rW3 + t * 8192,
        agg, edgeTiles);
    k_node<<<(nodeTiles + 7) / 8, 256, 0, stream>>>(nf_in, agg,
        W_lmix + t * 4096, W_scm + t * 4096, W_hid + t * 4096,
        g_sc + t * 320, W_prod + t * 960, W_node + t * 3200,
        node_types, nf_out, node_contrib, t, nodeTiles);
    k_edge<<<(edgeTiles + 3) / 4, 128, 0, stream>>>(eb, Y, edge_index, nf_out,
        W_er + t * 256, W_edge + t * 3200, edge_contrib, t, edgeTiles);
    float* tmp = nf_in; nf_in = nf_out; nf_out = tmp;
  }
  k_sum<<<(unsigned)(((size_t)NN * 100 + 255) / 256), 256, 0, stream>>>(node_contrib, node_sum, (size_t)NN * 100);
  k_sum<<<(unsigned)(((size_t)EE * 100 + 255) / 256), 256, 0, stream>>>(edge_contrib, edge_sum, (size_t)EE * 100);
}
